// FraudDetectionModel_75883482185773
// MI455X (gfx1250) — matrix operand order repaired and checked by reading, not compiled, not run on hardware
//
#include <hip/hip_runtime.h>
#include <hip/hip_bf16.h>
#include <math.h>

// ---------------- WMMA helpers (gfx1250, wave32) ----------------

typedef __attribute__((ext_vector_type(16))) _Float16 v16h;
typedef __attribute__((ext_vector_type(8)))  float    v8f;

#define DEV __device__ __forceinline__

DEV v8f wmma16(v16h a, v16h b, v8f c) {
  // D = A(16x32 f16) * B(32x16 f16) + C(16x16 f32)
  return __builtin_amdgcn_wmma_f32_16x16x32_f16(
      /*neg_a=*/false, a, /*neg_b=*/false, b,
      /*c_mod=*/(short)0, c, /*reuse_a=*/false, /*reuse_b=*/false);
}

// A fragment: 16x32 (MxK) f16 from row-major f32 matrix.
// lane<16: row=lane, elems 0..7 -> K=k0+0..7, elems 8..15 -> K=k0+16..23
// lane>=16: row=lane-16, elems 0..7 -> K=k0+8..15, elems 8..15 -> K=k0+24..31
DEV v16h load_a_f32(const float* base, int ld, int k0, int kmax, int lane) {
  const float* r = base + (size_t)(lane & 15) * ld;
  int kb = (lane >> 4) << 3;
  v16h a;
#pragma unroll
  for (int i = 0; i < 8; ++i) {
    int k1 = k0 + kb + i;
    int k2 = k1 + 16;
    a[i]     = (k1 < kmax) ? (_Float16)r[k1] : (_Float16)0.0f;
    a[i + 8] = (k2 < kmax) ? (_Float16)r[k2] : (_Float16)0.0f;
  }
  return a;
}

// B fragment: 32x16 (KxN) f16 from row-major f32 weight [K x ldw].
// lanes 0-15: col=lane, K = k0+0..7,16..23 ; lanes 16-31: col=lane-16, K = k0+8..15,24..31
DEV v16h load_b_f32(const float* W, int ldw, int n0, int k0, int kmax, int lane) {
  int col = n0 + (lane & 15);
  int kb  = (lane >> 4) << 3;
  v16h b;
#pragma unroll
  for (int i = 0; i < 16; ++i) {
    int k = k0 + kb + (i < 8 ? i : i + 8);
    b[i] = (k < kmax) ? (_Float16)W[(size_t)k * ldw + col] : (_Float16)0.0f;
  }
  return b;
}

// C/D fragment: 16x16 f32. vreg r, lane l -> row = r + 8*(l>=16), col = l&15
DEV void store_c_f32(float* out, int ld, int n0, v8f c, int lane) {
  int col = n0 + (lane & 15);
  int r0  = (lane >> 4) << 3;
#pragma unroll
  for (int r = 0; r < 8; ++r) out[(size_t)(r0 + r) * ld + col] = c[r];
}

DEV void wave_lds_fence() { asm volatile("s_wait_dscnt 0" ::: "memory"); }

DEV int fmax_key(float v) {           // order-preserving int key for float max
  int k = __float_as_int(v);
  return k >= 0 ? k : (k ^ 0x7fffffff);
}
DEV float fmax_unkey(int k) {
  k = k >= 0 ? k : (k ^ 0x7fffffff);
  return __int_as_float(k);
}

// ---------------- fill kernels ----------------

__global__ void k_fill_f(float* p, float v, int n) {
  int i = blockIdx.x * blockDim.x + threadIdx.x;
  if (i < n) p[i] = v;
}
__global__ void k_fill_i(int* p, int v, int n) {
  int i = blockIdx.x * blockDim.x + threadIdx.x;
  if (i < n) p[i] = v;
}

// ---------------- temporal encoder ----------------
// per wave: 16 nodes. For t=0..49: [16,10]@[10,64] WMMA, +b1, relu, accumulate.
// mean -> LDS -> [16,64]@[64,64] WMMA + b2 -> comb[:, 0:64]
__global__ __launch_bounds__(256) void k_temporal(
    const float* __restrict__ td, const float* __restrict__ tW1,
    const float* __restrict__ tb1, const float* __restrict__ tW2,
    const float* __restrict__ tb2, float* __restrict__ comb, int N) {
  __shared__ float lds[8][16][64];
  int lane = threadIdx.x & 31, w = threadIdx.x >> 5;
  int nb = (blockIdx.x * 8 + w) * 16;
  if (nb >= N) return;

  v16h bw1[4];
#pragma unroll
  for (int j = 0; j < 4; ++j) bw1[j] = load_b_f32(tW1, 64, j * 16, 0, 10, lane);
  float bias1[4];
#pragma unroll
  for (int j = 0; j < 4; ++j) bias1[j] = tb1[j * 16 + (lane & 15)];

  v8f macc[4];
#pragma unroll
  for (int j = 0; j < 4; ++j) macc[j] = (v8f){};

  for (int t = 0; t < 50; ++t) {
    const float* base = td + ((size_t)nb * 50 + t) * 10;
    v16h a = load_a_f32(base, 500, 0, 10, lane);
#pragma unroll
    for (int j = 0; j < 4; ++j) {
      v8f c = {};
      c = wmma16(a, bw1[j], c);
#pragma unroll
      for (int r = 0; r < 8; ++r) {
        float v = c[r] + bias1[j];
        macc[j][r] += v > 0.f ? v : 0.f;
      }
    }
  }
#pragma unroll
  for (int j = 0; j < 4; ++j) {
#pragma unroll
    for (int r = 0; r < 8; ++r) macc[j][r] *= 0.02f;   // 1/50
    store_c_f32(&lds[w][0][0], 64, j * 16, macc[j], lane);
  }
  wave_lds_fence();

  v16h a0 = load_a_f32(&lds[w][0][0], 64, 0, 64, lane);
  v16h a1 = load_a_f32(&lds[w][0][0], 64, 32, 64, lane);
#pragma unroll
  for (int j = 0; j < 4; ++j) {
    v8f c = {};
    c = wmma16(a0, load_b_f32(tW2, 64, j * 16, 0, 64, lane), c);
    c = wmma16(a1, load_b_f32(tW2, 64, j * 16, 32, 64, lane), c);
    float b2 = tb2[j * 16 + (lane & 15)];
#pragma unroll
    for (int r = 0; r < 8; ++r) c[r] += b2;
    store_c_f32(comb + (size_t)nb * 128, 128, j * 16, c, lane);
  }
}

// ---------------- GAT1 feature transform + attention coefficients ----------------
// h1 = x @ gW1 ([N,10]@[10,128]); a1s/a1d per (node, head) via LDS reduction.
__global__ __launch_bounds__(128) void k_gat1(
    const float* __restrict__ x, const float* __restrict__ gW1,
    const float* __restrict__ as, const float* __restrict__ ad,
    float* __restrict__ h1, float* __restrict__ a1s, float* __restrict__ a1d, int N) {
  __shared__ float lds[4][16][128];
  int lane = threadIdx.x & 31, w = threadIdx.x >> 5;
  int nb = (blockIdx.x * 4 + w) * 16;
  if (nb >= N) return;

  v16h a = load_a_f32(x + (size_t)nb * 10, 10, 0, 10, lane);
#pragma unroll
  for (int j = 0; j < 8; ++j) {
    v8f c = {};
    c = wmma16(a, load_b_f32(gW1, 128, j * 16, 0, 10, lane), c);
    store_c_f32(h1 + (size_t)nb * 128, 128, j * 16, c, lane);
    store_c_f32(&lds[w][0][0], 128, j * 16, c, lane);
  }
  wave_lds_fence();
#pragma unroll
  for (int i = 0; i < 4; ++i) {
    int tt = lane * 4 + i;        // 128 tasks: 16 nodes x 4 heads x {src,dst}
    int node = tt >> 3, rem = tt & 7;
    int head = rem >> 1, isd = rem & 1;
    const float* att = (isd ? ad : as) + head * 32;
    float s = 0.f;
#pragma unroll
    for (int cc = 0; cc < 32; ++cc) s += lds[w][node][head * 32 + cc] * att[cc];
    (isd ? a1d : a1s)[(size_t)(nb + node) * 4 + head] = s;
  }
}

// ---------------- GAT1 edge passes (4 heads) ----------------

DEV void edge_decode(const int* ei, int E, int N, int e, int& s, int& d) {
  s = (e < E) ? ei[e] : (e - E);
  d = (e < E) ? ei[E + e] : (e - E);
}

__global__ void k_edge_max1(const int* __restrict__ ei, int E, int N,
                            const float* __restrict__ a1s,
                            const float* __restrict__ a1d, int* __restrict__ m1) {
  int e = blockIdx.x * blockDim.x + threadIdx.x;
  if (e >= E + N) return;
  int s, d; edge_decode(ei, E, N, e, s, d);
  float4 vs = *(const float4*)(a1s + (size_t)s * 4);
  float4 vd = *(const float4*)(a1d + (size_t)d * 4);
  float ev[4] = {vs.x + vd.x, vs.y + vd.y, vs.z + vd.z, vs.w + vd.w};
#pragma unroll
  for (int h = 0; h < 4; ++h) {
    float v = ev[h] > 0.f ? ev[h] : 0.2f * ev[h];
    atomicMax(&m1[(size_t)d * 4 + h], fmax_key(v));
  }
}

__global__ void k_edge_exp1(const int* __restrict__ ei, int E, int N,
                            const float* __restrict__ a1s, const float* __restrict__ a1d,
                            const int* __restrict__ m1, float* __restrict__ d1,
                            float* __restrict__ wbuf) {
  int e = blockIdx.x * blockDim.x + threadIdx.x;
  if (e >= E + N) return;
  int s, d; edge_decode(ei, E, N, e, s, d);
  float4 vs = *(const float4*)(a1s + (size_t)s * 4);
  float4 vd = *(const float4*)(a1d + (size_t)d * 4);
  int4   mk = *(const int4*)(m1 + (size_t)d * 4);
  float ev[4] = {vs.x + vd.x, vs.y + vd.y, vs.z + vd.z, vs.w + vd.w};
  int   mv[4] = {mk.x, mk.y, mk.z, mk.w};
  float4 wv;
  float* wp = (float*)&wv;
#pragma unroll
  for (int h = 0; h < 4; ++h) {
    float v = ev[h] > 0.f ? ev[h] : 0.2f * ev[h];
    wp[h] = __expf(v - fmax_unkey(mv[h]));
    atomicAdd(&d1[(size_t)d * 4 + h], wp[h]);
  }
  *(float4*)(wbuf + (size_t)e * 4) = wv;
}

// one WAVE per edge: lanes cover the 128 channels -> coalesced gather + atomics
__global__ __launch_bounds__(256) void k_edge_aggr1(
    const int* __restrict__ ei, int E, int N,
    const float* __restrict__ wbuf, const float* __restrict__ d1,
    const float* __restrict__ h1, float* __restrict__ out1) {
  int lane = threadIdx.x & 31;
  int e = (blockIdx.x * blockDim.x + threadIdx.x) >> 5;
  if (e >= E + N) return;
  int s, d; edge_decode(ei, E, N, e, s, d);
  float4 wv = *(const float4*)(wbuf + (size_t)e * 4);   // wave-uniform
  float4 dv = *(const float4*)(d1 + (size_t)d * 4);
  float alpha[4] = {wv.x / dv.x, wv.y / dv.y, wv.z / dv.z, wv.w / dv.w};
  const float* hr = h1 + (size_t)s * 128;
  float* orow = out1 + (size_t)d * 128;
#pragma unroll
  for (int i = 0; i < 4; ++i) {                         // channel = lane + 32*i, head = i
    int ch = lane + 32 * i;
    atomicAdd(&orow[ch], hr[ch] * alpha[i]);
  }
}

__global__ void k_elu1(float* __restrict__ out1, const float* __restrict__ gb1, int n4) {
  int i = blockIdx.x * blockDim.x + threadIdx.x;        // n4 = N*32 float4's
  if (i >= n4) return;
  float4 v = ((const float4*)out1)[i];
  int c0 = (i & 31) * 4;
  float r[4] = {v.x + gb1[c0], v.y + gb1[c0 + 1], v.z + gb1[c0 + 2], v.w + gb1[c0 + 3]};
#pragma unroll
  for (int j = 0; j < 4; ++j) r[j] = r[j] > 0.f ? r[j] : (__expf(r[j]) - 1.f);
  ((float4*)out1)[i] = make_float4(r[0], r[1], r[2], r[3]);
}

// ---------------- GAT2 feature transform + attention coefficients ----------------
// h2 = elu_out @ gW2 ([N,128]@[128,64]); a2s/a2d (1 head) via LDS reduction.
__global__ __launch_bounds__(256) void k_gat2(
    const float* __restrict__ out1, const float* __restrict__ gW2,
    const float* __restrict__ as2, const float* __restrict__ ad2,
    float* __restrict__ h2, float* __restrict__ a2s, float* __restrict__ a2d, int N) {
  __shared__ float lds[8][16][64];
  int lane = threadIdx.x & 31, w = threadIdx.x >> 5;
  int nb = (blockIdx.x * 8 + w) * 16;
  if (nb >= N) return;

  const float* Ab = out1 + (size_t)nb * 128;
  v16h afr[4];
#pragma unroll
  for (int kc = 0; kc < 4; ++kc) afr[kc] = load_a_f32(Ab, 128, kc * 32, 128, lane);
#pragma unroll
  for (int j = 0; j < 4; ++j) {
    v8f c = {};
#pragma unroll
    for (int kc = 0; kc < 4; ++kc)
      c = wmma16(afr[kc], load_b_f32(gW2, 64, j * 16, kc * 32, 128, lane), c);
    store_c_f32(h2 + (size_t)nb * 64, 64, j * 16, c, lane);
    store_c_f32(&lds[w][0][0], 64, j * 16, c, lane);
  }
  wave_lds_fence();
  int node = lane >> 1, isd = lane & 1;   // 32 tasks: 16 nodes x {src,dst}
  const float* att = isd ? ad2 : as2;
  float s = 0.f;
#pragma unroll
  for (int cc = 0; cc < 64; ++cc) s += lds[w][node][cc] * att[cc];
  (isd ? a2d : a2s)[nb + node] = s;
}

// ---------------- GAT2 edge passes (1 head) ----------------

__global__ void k_edge_max2(const int* __restrict__ ei, int E, int N,
                            const float* __restrict__ a2s,
                            const float* __restrict__ a2d, int* __restrict__ m2) {
  int e = blockIdx.x * blockDim.x + threadIdx.x;
  if (e >= E + N) return;
  int s, d; edge_decode(ei, E, N, e, s, d);
  float v = a2s[s] + a2d[d];
  v = v > 0.f ? v : 0.2f * v;
  atomicMax(&m2[d], fmax_key(v));
}

__global__ void k_edge_exp2(const int* __restrict__ ei, int E, int N,
                            const float* __restrict__ a2s, const float* __restrict__ a2d,
                            const int* __restrict__ m2, float* __restrict__ d2,
                            float* __restrict__ wbuf) {
  int e = blockIdx.x * blockDim.x + threadIdx.x;
  if (e >= E + N) return;
  int s, d; edge_decode(ei, E, N, e, s, d);
  float v = a2s[s] + a2d[d];
  v = v > 0.f ? v : 0.2f * v;
  float wv = __expf(v - fmax_unkey(m2[d]));
  wbuf[e] = wv;
  atomicAdd(&d2[d], wv);
}

// one WAVE per edge: lanes cover the 64 channels
__global__ __launch_bounds__(256) void k_edge_aggr2(
    const int* __restrict__ ei, int E, int N,
    const float* __restrict__ wbuf, const float* __restrict__ d2,
    const float* __restrict__ h2, float* __restrict__ comb) {
  int lane = threadIdx.x & 31;
  int e = (blockIdx.x * blockDim.x + threadIdx.x) >> 5;
  if (e >= E + N) return;
  int s, d; edge_decode(ei, E, N, e, s, d);
  float alpha = wbuf[e] / d2[d];                        // wave-uniform
  const float* hr = h2 + (size_t)s * 64;
  float* orow = comb + (size_t)d * 128 + 64;
#pragma unroll
  for (int i = 0; i < 2; ++i) {
    int ch = lane + 32 * i;
    atomicAdd(&orow[ch], hr[ch] * alpha);
  }
}

__global__ void k_bias2(float* __restrict__ comb, const float* __restrict__ gb2, int n) {
  int i = blockIdx.x * blockDim.x + threadIdx.x;        // n = N*64
  if (i >= n) return;
  int node = i >> 6, c = i & 63;
  comb[(size_t)node * 128 + 64 + c] += gb2[c];
}

// ---------------- classifier: relu(comb@cW1+cb1)@cW2+cb2 -> sigmoid ----------------
__global__ __launch_bounds__(256) void k_cls(
    const float* __restrict__ comb, const float* __restrict__ cW1,
    const float* __restrict__ cb1, const float* __restrict__ cW2,
    const float* __restrict__ cb2, float* __restrict__ out, int N) {
  __shared__ float lds[8][16][64];
  int lane = threadIdx.x & 31, w = threadIdx.x >> 5;
  int nb = (blockIdx.x * 8 + w) * 16;
  if (nb >= N) return;

  const float* Ab = comb + (size_t)nb * 128;
  v16h afr[4];
#pragma unroll
  for (int kc = 0; kc < 4; ++kc) afr[kc] = load_a_f32(Ab, 128, kc * 32, 128, lane);
#pragma unroll
  for (int j = 0; j < 4; ++j) {
    v8f c = {};
#pragma unroll
    for (int kc = 0; kc < 4; ++kc)
      c = wmma16(afr[kc], load_b_f32(cW1, 64, j * 16, kc * 32, 128, lane), c);
    float b = cb1[j * 16 + (lane & 15)];
#pragma unroll
    for (int r = 0; r < 8; ++r) {
      float v = c[r] + b;
      c[r] = v > 0.f ? v : 0.f;
    }
    store_c_f32(&lds[w][0][0], 64, j * 16, c, lane);
  }
  wave_lds_fence();
  if (lane < 16) {
    float s = cb2[0];
#pragma unroll
    for (int k = 0; k < 64; ++k) s += lds[w][lane][k] * cW2[k];
    out[nb + lane] = 1.f / (1.f + __expf(-s));
  }
}

// ---------------- host launch ----------------

extern "C" void kernel_launch(void* const* d_in, const int* in_sizes, int n_in,
                              void* d_out, int out_size, void* d_ws, size_t ws_size,
                              hipStream_t stream) {
  const float* td   = (const float*)d_in[0];
  const float* x    = (const float*)d_in[1];
  const int*   ei   = (const int*)  d_in[2];
  const float* tW1  = (const float*)d_in[3];
  const float* tb1  = (const float*)d_in[4];
  const float* tW2  = (const float*)d_in[5];
  const float* tb2  = (const float*)d_in[6];
  const float* gW1  = (const float*)d_in[7];
  const float* ga1s = (const float*)d_in[8];
  const float* ga1d = (const float*)d_in[9];
  const float* gb1  = (const float*)d_in[10];
  const float* gW2  = (const float*)d_in[11];
  const float* ga2s = (const float*)d_in[12];
  const float* ga2d = (const float*)d_in[13];
  const float* gb2  = (const float*)d_in[14];
  const float* cW1  = (const float*)d_in[15];
  const float* cb1  = (const float*)d_in[16];
  const float* cW2  = (const float*)d_in[17];
  const float* cb2  = (const float*)d_in[18];
  float* out = (float*)d_out;

  const int N  = in_sizes[1] / 10;     // 100000
  const int E  = in_sizes[2] / 2;      // 1600000
  const int EL = E + N;                // edges + self-loops

  // workspace layout (floats)
  float* ws = (float*)d_ws;
  size_t o = 0;
  float* h1   = ws + o; o += (size_t)N * 128;
  float* out1 = ws + o; o += (size_t)N * 128;
  float* comb = ws + o; o += (size_t)N * 128;
  float* h2   = ws + o; o += (size_t)N * 64;
  float* a1s  = ws + o; o += (size_t)N * 4;
  float* a1d  = ws + o; o += (size_t)N * 4;
  float* d1   = ws + o; o += (size_t)N * 4;
  int*   m1   = (int*)(ws + o); o += (size_t)N * 4;
  float* a2s  = ws + o; o += (size_t)N;
  float* a2d  = ws + o; o += (size_t)N;
  float* d2   = ws + o; o += (size_t)N;
  int*   m2   = (int*)(ws + o); o += (size_t)N;
  float* wbuf = ws + o; o += (size_t)EL * 4;
  (void)ws_size; (void)n_in; (void)out_size;

  const int T = 256;
  const int NEG_INF_KEY = (int)0x80000000;

  // init accumulators
  k_fill_f<<<(N * 128 + T - 1) / T, T, 0, stream>>>(out1, 0.f, N * 128);
  k_fill_f<<<(N * 128 + T - 1) / T, T, 0, stream>>>(comb, 0.f, N * 128);
  k_fill_f<<<(N * 4 + T - 1) / T, T, 0, stream>>>(d1, 0.f, N * 4);
  k_fill_f<<<(N + T - 1) / T, T, 0, stream>>>(d2, 0.f, N);
  k_fill_i<<<(N * 4 + T - 1) / T, T, 0, stream>>>(m1, NEG_INF_KEY, N * 4);
  k_fill_i<<<(N + T - 1) / T, T, 0, stream>>>(m2, NEG_INF_KEY, N);

  const int tiles = (N + 15) / 16;
  const int blk8  = (tiles + 7) / 8;
  const int blk4  = (tiles + 3) / 4;

  // temporal encoder -> comb[:, 0:64]
  k_temporal<<<blk8, 256, 0, stream>>>(td, tW1, tb1, tW2, tb2, comb, N);

  // GAT layer 1
  k_gat1<<<blk4, 128, 0, stream>>>(x, gW1, ga1s, ga1d, h1, a1s, a1d, N);
  k_edge_max1<<<(EL + T - 1) / T, T, 0, stream>>>(ei, E, N, a1s, a1d, m1);
  k_edge_exp1<<<(EL + T - 1) / T, T, 0, stream>>>(ei, E, N, a1s, a1d, m1, d1, wbuf);
  k_edge_aggr1<<<(unsigned)(((size_t)EL * 32 + T - 1) / T), T, 0, stream>>>(
      ei, E, N, wbuf, d1, h1, out1);
  k_elu1<<<(N * 32 + T - 1) / T, T, 0, stream>>>(out1, gb1, N * 32);

  // GAT layer 2 -> comb[:, 64:128]
  k_gat2<<<blk8, 256, 0, stream>>>(out1, gW2, ga2s, ga2d, h2, a2s, a2d, N);
  k_edge_max2<<<(EL + T - 1) / T, T, 0, stream>>>(ei, E, N, a2s, a2d, m2);
  k_edge_exp2<<<(EL + T - 1) / T, T, 0, stream>>>(ei, E, N, a2s, a2d, m2, d2, wbuf);
  k_edge_aggr2<<<(unsigned)(((size_t)EL * 32 + T - 1) / T), T, 0, stream>>>(
      ei, E, N, wbuf, d2, h2, comb);
  k_bias2<<<(N * 64 + T - 1) / T, T, 0, stream>>>(comb, gb2, N * 64);

  // classifier
  k_cls<<<blk8, 256, 0, stream>>>(comb, cW1, cb1, cW2, cb2, out, N);
}
